// GRU_27822798144052
// MI455X (gfx1250) — matrix operand order repaired and checked by reading, not compiled, not run on hardware
//
#include <hip/hip_runtime.h>
#include <hip/hip_bf16.h>
#include <cstdint>
#include <cstddef>

// GRU on MI455X (gfx1250): bf16 WMMA (f32 accum), wave32, 16 waves/WG scan.
// B=128, S=2048, F=128, H=256, 3H=768.
#define B_  128
#define S_  2048
#define F_  128
#define H_  256
#define G_  768
#define GT_ 48       // 768/16 gate-column tiles
#define KB_IH 4      // F/32
#define KB_HH 8      // H/32
#define C_  32

typedef __attribute__((ext_vector_type(16))) __bf16 v16bf;
typedef __attribute__((ext_vector_type(8)))  __bf16 v8bf;
typedef __attribute__((ext_vector_type(8)))  float  v8f;

// A-matrix (16x32 bf16, MxK) fragment position per ISA table:
// lane = m + 16*h where h=(kk>>3)&1 ; vgpr v: kk in {2v+p+8h} (v<4) or {16+2(v-4)+p+8h}
__device__ __forceinline__ void a_frag_pos(int m, int kk, int& lane, int& j) {
    int h = (kk >> 3) & 1;
    int v = ((kk >> 4) << 2) + ((kk & 7) >> 1);
    j = 2 * v + (kk & 1);
    lane = m + (h << 4);
}

// ---------------------------------------------------------------------------
// Kernel 1: pack W_ih / W_hh (fp32 row-major [G,K]) into bf16 B-fragment
// layout: pack[gt][kblk][lane][16]. B striping (wave32): n=lane&15, h=lane>>4,
// k = kblk*32 + 8h + j (j<8), kblk*32 + 16 + 8h + (j-8) (j>=8).
// ---------------------------------------------------------------------------
__global__ void gru_pack_w(const float* __restrict__ Wih,
                           const float* __restrict__ Whh,
                           __bf16* __restrict__ ihp,
                           __bf16* __restrict__ hhp) {
    int t = blockIdx.x * blockDim.x + threadIdx.x;
    const int NIH = GT_ * KB_IH * 32;
    const int NHH = GT_ * KB_HH * 32;
    if (t < NIH) {
        int lane = t & 31, rest = t >> 5;
        int kblk = rest % KB_IH, gt = rest / KB_IH;
        int g  = gt * 16 + (lane & 15);
        int k0 = kblk * 32 + ((lane >> 4) << 3);
        __bf16* dst = ihp + ((size_t)(gt * KB_IH + kblk) * 32 + lane) * 16;
        const float* src = Wih + (size_t)g * F_ + k0;
        #pragma unroll
        for (int j = 0; j < 16; ++j) dst[j] = (__bf16)src[j + ((j >> 3) << 3)];
    } else if (t < NIH + NHH) {
        t -= NIH;
        int lane = t & 31, rest = t >> 5;
        int kblk = rest % KB_HH, gt = rest / KB_HH;
        int g  = gt * 16 + (lane & 15);
        int k0 = kblk * 32 + ((lane >> 4) << 3);
        __bf16* dst = hhp + ((size_t)(gt * KB_HH + kblk) * 32 + lane) * 16;
        const float* src = Whh + (size_t)g * H_ + k0;
        #pragma unroll
        for (int j = 0; j < 16; ++j) dst[j] = (__bf16)src[j + ((j >> 3) << 3)];
    }
}

// ---------------------------------------------------------------------------
// Kernel 2: gi = x @ W_ih^T + b_ih  (51.5 GFLOP, fully parallel WMMA GEMM)
// Stored bf16 in C-layout fragments: gip[s][mt][gt][lane][8].
// Grid (S, 48), 256 threads (8 waves; wave = batch tile mt).
// ---------------------------------------------------------------------------
__global__ void __launch_bounds__(256)
gru_gi_gemm(const float* __restrict__ x, const __bf16* __restrict__ ihp,
            const float* __restrict__ b_ih, __bf16* __restrict__ gip) {
    __shared__ __bf16 xfrag[8][KB_IH][32][16];   // 32 KB: x tile in A-frag layout
    const int s = blockIdx.x, gt = blockIdx.y;
    const int tid = threadIdx.x, lane = tid & 31, wave = tid >> 5;

    // Stage x[:, s, :] (128x128 f32) -> bf16 A-fragments (coalesced reads)
    for (int e = tid; e < B_ * F_; e += 256) {
        int bl = e >> 7, k = e & 127;
        float xv = x[((size_t)bl * S_ + s) * F_ + k];
        int mt = bl >> 4, m = bl & 15;
        int kblk = k >> 5, kk = k & 31;
        int ln, j; a_frag_pos(m, kk, ln, j);
        xfrag[mt][kblk][ln][j] = (__bf16)xv;
    }
    __syncthreads();

    const int mt = wave;
    float bias = b_ih[gt * 16 + (lane & 15)];   // column-only dependence in C layout
    v8f c;
    #pragma unroll
    for (int i = 0; i < 8; ++i) c[i] = bias;

    #pragma unroll
    for (int kblk = 0; kblk < KB_IH; ++kblk) {
        v16bf a = *(const v16bf*)&xfrag[mt][kblk][lane][0];
        v16bf b = *(const v16bf*)(ihp + ((size_t)(gt * KB_IH + kblk) * 32 + lane) * 16);
        c = __builtin_amdgcn_wmma_f32_16x16x32_bf16(false, a, false, b,
                                                    (short)0, c, false, false);
    }

    v8bf o;
    #pragma unroll
    for (int i = 0; i < 8; ++i) o[i] = (__bf16)c[i];
    *(v8bf*)(gip + (((size_t)s * 8 + mt) * GT_ + gt) * 32 * 8 + (size_t)lane * 8) = o;
}

// ---------------------------------------------------------------------------
// Kernel 3: persistent recurrence. 8 independent WGs (16 batch rows each),
// 512 threads = 16 waves; wave w owns hidden tile hi=w and its r/z/n gates.
// h ping-pongs through LDS in bf16 A-fragment layout; one barrier per step.
// r/z gate weight fragments live in VGPRs (128 regs) for the whole scan;
// only n-gate weights stream from L2 (cuts per-step L2 traffic by 2/3).
// Per wave per step: 24 v_wmma_f32_16x16x32_bf16 (K=256, 3 gates).
// ---------------------------------------------------------------------------
__global__ void __launch_bounds__(512)
gru_scan(const __bf16* __restrict__ gip, const __bf16* __restrict__ hhp,
         const float* __restrict__ b_hh, float* __restrict__ h_out) {
    __shared__ __bf16 hfrag[2][KB_HH][32][16];   // 16 KB double-buffered h
    const int tid = threadIdx.x, lane = tid & 31, wave = tid >> 5;
    const int wg = blockIdx.x;      // batch tile (rows wg*16 .. +16)
    const int hi = wave;            // hidden column tile 0..15

    for (int e = tid; e < 2 * KB_HH * 32 * 16; e += 512)
        ((__bf16*)&hfrag[0][0][0][0])[e] = (__bf16)0.0f;

    const int col = hi * 16 + (lane & 15);       // hidden column owned (C layout n)
    const float bhr = b_hh[col], bhz = b_hh[H_ + col], bhn = b_hh[2 * H_ + col];
    v8f hreg;
    #pragma unroll
    for (int i = 0; i < 8; ++i) hreg[i] = 0.0f;

    // Register-resident B fragments for r and z gates (16 x v16bf = 128 VGPRs)
    v16bf brw[KB_HH], bzw[KB_HH];
    #pragma unroll
    for (int kblk = 0; kblk < KB_HH; ++kblk) {
        brw[kblk] = *(const v16bf*)(hhp + ((size_t)((hi     ) * KB_HH + kblk) * 32 + lane) * 16);
        bzw[kblk] = *(const v16bf*)(hhp + ((size_t)((hi + 16) * KB_HH + kblk) * 32 + lane) * 16);
    }

    // Precompute LDS write position for h_new column `col` in A-frag layout
    const int kblk_w = col >> 5, kk = col & 31;
    const int lh = (kk >> 3) & 1;
    const int jw = 2 * (((kk >> 4) << 2) + ((kk & 7) >> 1)) + (kk & 1);
    __syncthreads();

    for (int t = 0; t < S_; ++t) {
        const int cur = t & 1, nxt = cur ^ 1;
        const size_t gbase = ((size_t)t * 8 + wg) * GT_;

        v8bf gir = *(const v8bf*)(gip + (gbase + hi)      * 256 + (size_t)lane * 8);
        v8bf giz = *(const v8bf*)(gip + (gbase + hi + 16) * 256 + (size_t)lane * 8);
        v8f cr, cz, cn;
        #pragma unroll
        for (int i = 0; i < 8; ++i) {
            cr[i] = (float)gir[i] + bhr;
            cz[i] = (float)giz[i] + bhz;
            cn[i] = 0.0f;                 // h-side n kept separate: n=tanh(i_n+r*(gh_n+b))
        }

        #pragma unroll
        for (int kblk = 0; kblk < KB_HH; ++kblk) {
            v16bf a  = *(const v16bf*)&hfrag[cur][kblk][lane][0];
            v16bf bn = *(const v16bf*)(hhp + ((size_t)((hi + 32) * KB_HH + kblk) * 32 + lane) * 16);
            cr = __builtin_amdgcn_wmma_f32_16x16x32_bf16(false, a, false, brw[kblk], (short)0, cr, false, false);
            cz = __builtin_amdgcn_wmma_f32_16x16x32_bf16(false, a, false, bzw[kblk], (short)0, cz, false, false);
            cn = __builtin_amdgcn_wmma_f32_16x16x32_bf16(false, a, false, bn,        (short)0, cn, false, false);
        }

        v8bf gin = *(const v8bf*)(gip + (gbase + hi + 32) * 256 + (size_t)lane * 8);

        if (t + 1 < S_) {   // hide next step's gi behind gate math (global_prefetch_b8)
            const size_t nbase = ((size_t)(t + 1) * 8 + wg) * GT_;
            __builtin_prefetch((const char*)(gip + (nbase + hi)      * 256), 0, 1);
            __builtin_prefetch((const char*)(gip + (nbase + hi + 16) * 256), 0, 1);
            __builtin_prefetch((const char*)(gip + (nbase + hi + 32) * 256), 0, 1);
        }

        #pragma unroll
        for (int i = 0; i < 8; ++i) {
            float r   = 1.0f / (1.0f + __expf(-cr[i]));
            float z   = 1.0f / (1.0f + __expf(-cz[i]));
            float n   = tanhf((float)gin[i] + r * (cn[i] + bhn));
            float hn  = (1.0f - z) * n + z * hreg[i];
            hreg[i] = hn;
            int m = i + ((lane >> 4) << 3);          // C-layout row
            hfrag[nxt][kblk_w][m + (lh << 4)][jw] = (__bf16)hn;
        }
        __syncthreads();   // writes to nxt visible before anyone reads it at t+1
    }

    #pragma unroll
    for (int i = 0; i < 8; ++i) {
        int m = i + ((lane >> 4) << 3);
        h_out[(size_t)(wg * 16 + m) * H_ + col] = hreg[i];
    }
}

// ---------------------------------------------------------------------------
// Kernel 4: out = h_last @ fc_w^T + fc_b   (128x32, trivial)
// ---------------------------------------------------------------------------
__global__ void gru_head(const float* __restrict__ h, const float* __restrict__ fcw,
                         const float* __restrict__ fcb, float* __restrict__ out) {
    int idx = blockIdx.x * blockDim.x + threadIdx.x;
    if (idx >= B_ * C_) return;
    int b = idx >> 5, c = idx & 31;
    float acc = fcb[c];
    const float* hp = h + (size_t)b * H_;
    const float* wp = fcw + (size_t)c * H_;
    for (int k = 0; k < H_; ++k) acc += hp[k] * wp[k];
    out[idx] = acc;
}

// ---------------------------------------------------------------------------
extern "C" void kernel_launch(void* const* d_in, const int* in_sizes, int n_in,
                              void* d_out, int out_size, void* d_ws, size_t ws_size,
                              hipStream_t stream) {
    const float* x    = (const float*)d_in[0];
    const float* Wih  = (const float*)d_in[1];
    const float* Whh  = (const float*)d_in[2];
    const float* b_ih = (const float*)d_in[3];
    const float* b_hh = (const float*)d_in[4];
    const float* fcw  = (const float*)d_in[5];
    const float* fcb  = (const float*)d_in[6];
    float* out = (float*)d_out;

    // Workspace carve (bytes), 256-aligned. Total ~403.4 MB.
    char* ws = (char*)d_ws;
    size_t off = 0;
    __bf16* ihp = (__bf16*)(ws + off); off += (size_t)GT_ * KB_IH * 32 * 16 * 2;      // 196,608 B
    off = (off + 255) & ~(size_t)255;
    __bf16* hhp = (__bf16*)(ws + off); off += (size_t)GT_ * KB_HH * 32 * 16 * 2;      // 393,216 B
    off = (off + 255) & ~(size_t)255;
    float*  h_last = (float*)(ws + off); off += (size_t)B_ * H_ * 4;                  // 131,072 B
    off = (off + 255) & ~(size_t)255;
    __bf16* gip = (__bf16*)(ws + off);   // S*8*48*256 bf16 = 402,653,184 B
    (void)ws_size; (void)in_sizes; (void)n_in; (void)out_size;

    gru_pack_w<<<dim3((GT_ * (KB_IH + KB_HH) * 32 + 255) / 256), dim3(256), 0, stream>>>(
        Wih, Whh, ihp, hhp);

    gru_gi_gemm<<<dim3(S_, GT_), dim3(256), 0, stream>>>(x, ihp, b_ih, gip);

    gru_scan<<<dim3(8), dim3(512), 0, stream>>>(gip, hhp, b_hh, h_last);

    gru_head<<<dim3((B_ * C_ + 255) / 256), dim3(256), 0, stream>>>(h_last, fcw, fcb, out);
}
